// MineralFusion_11802570130087
// MI455X (gfx1250) — matrix operand order repaired and checked by reading, not compiled, not run on hardware
//
#include <hip/hip_runtime.h>
#include <math.h>

#define BB 32
#define CC 256
#define HH 56
#define WW 56
#define HWX (HH*WW)          // 3136
#define KSEL 30
#define XR 62                // 56 + 2*3 halo (7x7 needs 3)
#define FR 58                // 56 + 2*1 halo (score 3x3 on fused)
#define NTHREADS 256

#if defined(__has_builtin)
#  if __has_builtin(__builtin_amdgcn_global_load_async_to_lds_b32) && \
      __has_builtin(__builtin_amdgcn_s_wait_asynccnt)
#    define USE_ASYNC_LDS 1
#  endif
#endif
#ifndef USE_ASYNC_LDS
#  define USE_ASYNC_LDS 0
#endif

typedef __attribute__((ext_vector_type(16))) __bf16 v16bf;
typedef __attribute__((ext_vector_type(8)))  float  v8f;

#if USE_ASYNC_LDS
typedef __attribute__((address_space(1))) int g_i32;   // global int
typedef __attribute__((address_space(3))) int l_i32;   // LDS int
#endif

// ---------------------------------------------------------------------------
// Kernel 1: per-(b,c) plane. dwconv3+dwconv5 -> fused, score conv, top-K mask,
// dwconv7, residual mix, plane mean. Everything LDS-resident (one HBM read of
// x, one HBM write of y -> bandwidth-optimal on 23.3 TB/s HBM). x-plane is
// staged with gfx1250 async global->LDS copies when available.
// ---------------------------------------------------------------------------
__global__ __launch_bounds__(NTHREADS)
void fusion_plane_kernel(const float* __restrict__ x,
                         const float* __restrict__ w1, const float* __restrict__ b1,
                         const float* __restrict__ w2, const float* __restrict__ b2,
                         const float* __restrict__ w3, const float* __restrict__ b3,
                         const float* __restrict__ wsc, const float* __restrict__ bs,
                         const float* __restrict__ alpha,
                         float* __restrict__ y_out, float* __restrict__ g_out)
{
    __shared__ float    sx[XR*XR];               // 3844 f: x plane + halo 3
    __shared__ float    sf[FR*FR];               // 3364 f: fused (c1+c2) + halo 1
    __shared__ float    ss[HWX];                 // 3136 f: scores
    __shared__ unsigned smask[(HWX+31)/32];      // top-K bitmask
    __shared__ float    rv[NTHREADS];
    __shared__ int      ri[NTHREADS];
    __shared__ float    wsh[96];                 // w1(9) w2(25) w3(49) ws(9) b(4)

    const int tid = threadIdx.x;
    const int bid = blockIdx.x;                  // = b*C + c
    const int ch  = bid % CC;
    const float* xp = x + (size_t)bid * HWX;

    __builtin_prefetch(xp, 0, 0);                // global_prefetch_b8

    if (tid < 96) {
        float v;
        if (tid < 9)        v = w1 [ch*9  + tid];
        else if (tid < 34)  v = w2 [ch*25 + (tid-9)];
        else if (tid < 83)  v = w3 [ch*49 + (tid-34)];
        else if (tid < 92)  v = wsc[ch*9  + (tid-83)];
        else if (tid == 92) v = b1[ch];
        else if (tid == 93) v = b2[ch];
        else if (tid == 94) v = b3[ch];
        else                v = bs[ch];
        wsh[tid] = v;
    }
    for (int i = tid; i < (HWX+31)/32; i += NTHREADS) smask[i] = 0u;

    // stage x plane with halo-3, zero padded ('SAME').
    // In-range elements: async global->LDS DMA (no VGPR round-trip, ASYNCcnt).
    for (int i = tid; i < XR*XR; i += NTHREADS) {
        int r = i / XR - 3, cc2 = i % XR - 3;
        bool inb = (r >= 0 && r < HH && cc2 >= 0 && cc2 < WW);
#if USE_ASYNC_LDS
        if (inb) {
            __builtin_amdgcn_global_load_async_to_lds_b32(
                (g_i32*)(xp + r*WW + cc2),
                (l_i32*)&sx[i],
                0, 0);
        } else {
            sx[i] = 0.f;
        }
#else
        sx[i] = inb ? xp[r*WW + cc2] : 0.f;
#endif
    }
#if USE_ASYNC_LDS
    __builtin_amdgcn_s_wait_asynccnt(0);         // s_wait_asynccnt 0
#endif
    __syncthreads();

    const float b1v = wsh[92], b2v = wsh[93], b3v = wsh[94], bsv = wsh[95];

    // fused = c1 + c2 over extended 58x58 region (halo 1 for score conv)
    for (int i = tid; i < FR*FR; i += NTHREADS) {
        int r = i / FR - 1, cc2 = i % FR - 1;    // plane coords -1..56
        const float* base = &sx[(r+3)*XR + (cc2+3)];
        float acc = b1v + b2v;
        #pragma unroll
        for (int dy = -1; dy <= 1; ++dy)
            #pragma unroll
            for (int dx = -1; dx <= 1; ++dx)
                acc += wsh[(dy+1)*3 + (dx+1)] * base[dy*XR + dx];
        #pragma unroll
        for (int dy = -2; dy <= 2; ++dy)
            #pragma unroll
            for (int dx = -2; dx <= 2; ++dx)
                acc += wsh[9 + (dy+2)*5 + (dx+2)] * base[dy*XR + dx];
        sf[i] = acc;
    }
    __syncthreads();

    // scores: 3x3 dwconv over fused
    for (int i = tid; i < HWX; i += NTHREADS) {
        int r = i / WW, cc2 = i % WW;
        const float* base = &sf[(r+1)*FR + (cc2+1)];
        float acc = bsv;
        #pragma unroll
        for (int dy = -1; dy <= 1; ++dy)
            #pragma unroll
            for (int dx = -1; dx <= 1; ++dx)
                acc += wsh[83 + (dy+1)*3 + (dx+1)] * base[dy*FR + dx];
        ss[i] = acc;
    }
    __syncthreads();

    // top-K: K rounds of block argmax (ties -> smallest index, like lax.top_k)
    for (int it = 0; it < KSEL; ++it) {
        float lv = -3.4e38f; int li = 0;
        for (int i = tid; i < HWX; i += NTHREADS) {
            float v = ss[i];
            if (v > lv) { lv = v; li = i; }
        }
        rv[tid] = lv; ri[tid] = li;
        __syncthreads();
        for (int s = NTHREADS/2; s > 0; s >>= 1) {
            if (tid < s) {
                float v2 = rv[tid+s]; int i2 = ri[tid+s];
                if (v2 > rv[tid] || (v2 == rv[tid] && i2 < ri[tid])) {
                    rv[tid] = v2; ri[tid] = i2;
                }
            }
            __syncthreads();
        }
        if (tid == 0) {
            int idx = ri[0];
            smask[idx >> 5] |= (1u << (idx & 31));
            ss[idx] = -3.4e38f;
        }
        __syncthreads();
    }

    // y = a*fused*mask + (1-a)*c3 + x ; accumulate plane sum for SE mean
    const float av = 1.f / (1.f + expf(-alpha[0]));
    float lsum = 0.f;
    float* yp = y_out + (size_t)bid * HWX;
    for (int i = tid; i < HWX; i += NTHREADS) {
        int r = i / WW, cc2 = i % WW;
        const float* base = &sx[(r+3)*XR + (cc2+3)];
        float c3 = b3v;
        #pragma unroll
        for (int dy = -3; dy <= 3; ++dy)
            #pragma unroll
            for (int dx = -3; dx <= 3; ++dx)
                c3 += wsh[34 + (dy+3)*7 + (dx+3)] * base[dy*XR + dx];
        float fv = sf[(r+1)*FR + (cc2+1)];
        float m  = (float)((smask[i >> 5] >> (i & 31)) & 1u);
        float yv = av * fv * m + (1.f - av) * c3 + base[0];
        yp[i] = yv;
        lsum += yv;
    }
    __syncthreads();
    rv[tid] = lsum;
    __syncthreads();
    for (int s = NTHREADS/2; s > 0; s >>= 1) {
        if (tid < s) rv[tid] += rv[tid+s];
        __syncthreads();
    }
    if (tid == 0) g_out[bid] = rv[0] * (1.f / (float)HWX);
}

// ---------------------------------------------------------------------------
// Kernel 2: SE block via v_wmma_f32_16x16x32_bf16.
//   GEMM1: hbott(32x16) = relu(g(32x256) @ se_w1^T)   -> 2 tiles x 8 k-steps
//   GEMM2: gate(32x256) = sigmoid(hbott @ se_w2^T)    -> 32 tiles, K padded
// Fragment layouts used here (contiguous K in A and B, not the ISA table order):
//   A 16x32 bf16: lane<16 row=lane, K=0..15; lane>=16 K=16..31
//   B 32x16 bf16: lane<16 col=lane, K=0..15; lane>=16 K=16..31
//   C/D f32: vgpr r, lane -> (M = r + 8*(lane>=16), N = lane&15)
// ---------------------------------------------------------------------------
__global__ __launch_bounds__(256)
void se_gemm_kernel(const float* __restrict__ g,      // (32,256)
                    const float* __restrict__ se_w1,  // (16,256)
                    const float* __restrict__ se_w2,  // (256,16)
                    float* __restrict__ gate)         // (32,256)
{
    __shared__ float hb[32*16];
    const int tid  = threadIdx.x;
    const int wid  = tid >> 5;
    const int lane = tid & 31;
    const int hi   = (lane >= 16) ? 1 : 0;
    const int l15  = lane & 15;

    if (wid < 2) {                                // wave-uniform: EXEC stays all-1
        const int mtile = wid;
        v8f acc = {};
        for (int ks = 0; ks < 8; ++ks) {
            const int kbase = ks * 32;
            v16bf af, bfv;
            const int arow = mtile*16 + l15;
            #pragma unroll
            for (int j = 0; j < 16; ++j) {
                int k = kbase + (hi ? 16 : 0) + j;
                af[j] = (__bf16)g[arow*256 + k];
            }
            const int kb = kbase + (hi ? 16 : 0);
            #pragma unroll
            for (int j = 0; j < 16; ++j)
                bfv[j] = (__bf16)se_w1[l15*256 + kb + j];
            acc = __builtin_amdgcn_wmma_f32_16x16x32_bf16(
                      false, af, false, bfv, (short)0, acc, false, false);
        }
        #pragma unroll
        for (int r = 0; r < 8; ++r) {
            float v = acc[r] > 0.f ? acc[r] : 0.f;      // relu
            hb[(mtile*16 + r + hi*8)*16 + l15] = v;
        }
    }
    __syncthreads();

    #pragma unroll
    for (int tt = 0; tt < 4; ++tt) {
        const int tile  = wid * 4 + tt;           // 32 tiles over 8 waves
        const int mtile = tile >> 4;              // 0..1
        const int ntile = tile & 15;              // 0..15
        v16bf af, bfv;
        const int arow = mtile*16 + l15;
        #pragma unroll
        for (int j = 0; j < 16; ++j) {
            int k = (hi ? 16 : 0) + j;
            af[j] = (k < 16) ? (__bf16)hb[arow*16 + k] : (__bf16)0.f;
        }
        #pragma unroll
        for (int j = 0; j < 16; ++j) {
            int k = (hi ? 16 : 0) + j;            // K 16..31 -> zero pad
            bfv[j] = (k < 16) ? (__bf16)se_w2[(ntile*16 + l15)*16 + k] : (__bf16)0.f;
        }
        v8f acc = {};
        acc = __builtin_amdgcn_wmma_f32_16x16x32_bf16(
                  false, af, false, bfv, (short)0, acc, false, false);
        #pragma unroll
        for (int r = 0; r < 8; ++r) {
            float s = 1.f / (1.f + expf(-acc[r]));
            gate[(mtile*16 + r + hi*8)*256 + ntile*16 + l15] = s;
        }
    }
}

// ---------------------------------------------------------------------------
// Kernel 3: out = y * (1 + gate[b,c]) in place, float4 vectorized.
// ---------------------------------------------------------------------------
__global__ __launch_bounds__(256)
void se_scale_kernel(float* __restrict__ y, const float* __restrict__ gate)
{
    const int i = blockIdx.x * blockDim.x + threadIdx.x;   // float4 index
    const int plane = (i * 4) / HWX;                       // HWX % 4 == 0
    float4* p = reinterpret_cast<float4*>(y) + i;
    const float s = 1.f + gate[plane];
    float4 v = *p;
    v.x *= s; v.y *= s; v.z *= s; v.w *= s;
    *p = v;
}

extern "C" void kernel_launch(void* const* d_in, const int* in_sizes, int n_in,
                              void* d_out, int out_size, void* d_ws, size_t ws_size,
                              hipStream_t stream)
{
    const float* x     = (const float*)d_in[0];
    const float* w1    = (const float*)d_in[1];
    const float* b1    = (const float*)d_in[2];
    const float* w2    = (const float*)d_in[3];
    const float* b2    = (const float*)d_in[4];
    const float* w3    = (const float*)d_in[5];
    const float* b3    = (const float*)d_in[6];
    const float* wsc   = (const float*)d_in[7];
    const float* bs    = (const float*)d_in[8];
    const float* se_w1 = (const float*)d_in[9];
    const float* se_w2 = (const float*)d_in[10];
    const float* alpha = (const float*)d_in[11];

    float* out  = (float*)d_out;
    float* g    = (float*)d_ws;          // (32,256) plane means
    float* gate = g + BB*CC;             // (32,256) SE gates

    fusion_plane_kernel<<<BB*CC, NTHREADS, 0, stream>>>(
        x, w1, b1, w2, b2, w3, b3, wsc, bs, alpha, out, g);

    se_gemm_kernel<<<1, 256, 0, stream>>>(g, se_w1, se_w2, gate);

    const int n4 = (BB*CC*HWX) / 4;      // 6,422,528 -> exactly 25088 blocks
    se_scale_kernel<<<n4/256, 256, 0, stream>>>(out, gate);
}
